// ETC_23184233464578
// MI455X (gfx1250) — compile-verified, matrix operand order checked by reading, not run on hardware
//
#include <hip/hip_runtime.h>
#include <hip/hip_bf16.h>

// ---- problem constants (match reference) ----
constexpr int B_    = 2;
constexpr int SL_   = 2048;
constexpr int RATIO_= 16;
constexpr int SG_   = SL_ / RATIO_;   // 128
constexpr int S_    = SL_ + SG_;      // 2176
constexpr int D_    = 512;
constexpr int H_    = 8;
constexpr int F_    = 2048;
constexpr int L_    = 2;
constexpr int NC_   = 8;
constexpr int HD_   = D_ / H_;        // 64
constexpr int BS_   = B_ * S_;        // 4352 rows (multiple of 64)

// LDS tile row stride in floats: 144 B = 9*16 B keeps 16B chunk alignment,
// and 16 consecutive rows land on 16 distinct 256B bank phases -> no LDS
// bank conflicts for the half-wave ds_load_b128 operand fetches.
constexpr int LDST_ = 36;

typedef __attribute__((ext_vector_type(16))) _Float16 v16h;
typedef __attribute__((ext_vector_type(8)))  float    v8f;

// -------------------------------------------------------------------------
// Fill one lane's WMMA f16 16x16x32 operand from an LDS tile row (fp32).
// Per-lane K layout (ISA 7.12.2, 16-bit A 16x32): lanes 0-15 hold
// K = {0..7, 16..23}, lanes 16-31 hold K = {8..15, 24..31}.
// -------------------------------------------------------------------------
__device__ __forceinline__ v16h lds_row_to_op(const float* __restrict__ p) {
    v16h r;
#pragma unroll
    for (int j = 0; j < 8; ++j) r[j]     = (_Float16)p[j];
#pragma unroll
    for (int j = 0; j < 8; ++j) r[8 + j] = (_Float16)p[16 + j];
    return r;
}

// Issue async-DMA of one 64x32 fp32 slab pair (A, W) into LDS buffer `buf`.
// 512 16-byte chunks per slab, 128 threads -> 4 chunks each, 8 async ops/wave.
__device__ __forceinline__ void stage_tiles(const float* __restrict__ A,
                                            const float* __restrict__ W,
                                            int K, int m0, int n0, int k0,
                                            int tid, unsigned ldsA, unsigned ldsW) {
#pragma unroll
    for (int j = 0; j < 4; ++j) {
        const int c   = tid + 128 * j;         // 0..511
        const int row = c >> 3;                // tile row 0..63
        const int col = (c & 7) << 2;          // float col 0,4,..,28
        const unsigned lo = (unsigned)((row * LDST_ + col) * 4);
        const unsigned long long ga =
            (unsigned long long)(uintptr_t)(A + (size_t)(m0 + row) * K + k0 + col);
        const unsigned long long gw =
            (unsigned long long)(uintptr_t)(W + (size_t)(n0 + row) * K + k0 + col);
        asm volatile("global_load_async_to_lds_b128 %0, %1, off"
                     :: "v"(ldsA + lo), "v"(ga) : "memory");
        asm volatile("global_load_async_to_lds_b128 %0, %1, off"
                     :: "v"(ldsW + lo), "v"(gw) : "memory");
    }
}

// -------------------------------------------------------------------------
// C = A(MxK) @ W(NxK)^T + bias, optional ReLU.
// Workgroup = 4 waves computes a 64x64 tile.  K-slabs are double-buffered
// in LDS via global_load_async_to_lds_b128 (ASYNCcnt): the next slab's DMA
// runs while the current slab feeds 4 WMMAs per wave.  Async loads complete
// in order, so s_wait_asynccnt <= 8 retires exactly the current batch.
// -------------------------------------------------------------------------
template <bool RELU>
__global__ __launch_bounds__(128)
void gemm_wmma(const float* __restrict__ A,
               const float* __restrict__ W,
               const float* __restrict__ bias,
               float* __restrict__ C,
               int M, int N, int K) {
    __shared__ float tA[2][64 * LDST_];        // 2 x 9 KB
    __shared__ float tW[2][64 * LDST_];        // 2 x 9 KB

    const int tid     = threadIdx.x;           // 0..127
    const int wave    = tid >> 5;              // 0..3  -> M sub-tile
    const int lane    = tid & 31;
    const int halfsel = lane >> 4;             // K sub-block select
    const int r       = lane & 15;

    const int m0 = blockIdx.x * 64;
    const int n0 = blockIdx.y * 64;

    // LDS byte offsets: generic shared address keeps LDS offset in [31:0]
    const unsigned ldsA = (unsigned)(uintptr_t)&tA[0][0];
    const unsigned ldsW = (unsigned)(uintptr_t)&tW[0][0];
    const unsigned bufB = (unsigned)(64 * LDST_ * 4);   // bytes per buffer

    v8f acc[4] = {v8f{}, v8f{}, v8f{}, v8f{}};

    const int NT = K >> 5;                     // K/32 slabs
    stage_tiles(A, W, K, m0, n0, 0, tid, ldsA, ldsW);

    for (int kt = 0; kt < NT; ++kt) {
        const int cur = kt & 1;
        if (kt + 1 < NT) {
            // prefetch next slab into the other buffer (WAR covered by the
            // end-of-iteration barrier of the previous loop trip)
            stage_tiles(A, W, K, m0, n0, (kt + 1) << 5, tid,
                        ldsA + (cur ^ 1) * bufB, ldsW + (cur ^ 1) * bufB);
            asm volatile("s_wait_asynccnt 0x8" ::: "memory");
        } else {
            asm volatile("s_wait_asynccnt 0x0" ::: "memory");
        }
        __syncthreads();                       // current slab visible to all

        const int kb = halfsel * 8;
        v16h a = lds_row_to_op(&tA[cur][(wave * 16 + r) * LDST_ + kb]);
#pragma unroll
        for (int nt = 0; nt < 4; ++nt) {
            v16h b = lds_row_to_op(&tW[cur][(nt * 16 + r) * LDST_ + kb]);
            acc[nt] = __builtin_amdgcn_wmma_f32_16x16x32_f16(
                false, a, false, b, (short)0, acc[nt], false, false);
        }
        __syncthreads();                       // done reading before overwrite
    }

    // ---- epilogue: bias (+ReLU), store -----------------------------------
#pragma unroll
    for (int nt = 0; nt < 4; ++nt) {
        const int col = n0 + nt * 16 + r;
        const float bv = bias[col];
#pragma unroll
        for (int i = 0; i < 8; ++i) {
            // C/D layout: VGPR i -> row i (lanes 0-15) / row i+8 (lanes 16-31)
            const int row = m0 + wave * 16 + i + halfsel * 8;
            float v = acc[nt][i] + bv;
            if (RELU) v = fmaxf(v, 0.0f);
            C[(size_t)row * N + col] = v;
        }
    }
}

// -------------------------------------------------------------------------
// Pad masks: pad_long (B,SL), pad_glob (B,SG)
// -------------------------------------------------------------------------
__global__ void mask_kernel(const int* __restrict__ tok,
                            int* __restrict__ pad_long,
                            int* __restrict__ pad_glob) {
    int g = blockIdx.x * blockDim.x + threadIdx.x;
    if (g >= B_ * SG_) return;
    int b = g / SG_, s = g % SG_;
    int allpad = 1;
    for (int j = 0; j < RATIO_; ++j) {
        int idx = b * SL_ + s * RATIO_ + j;
        int p = (tok[idx] == 0);
        pad_long[idx] = p;
        allpad &= p;
    }
    pad_glob[g] = allpad;
}

// Embedding gather + global-token broadcast -> x (B,S,D)
__global__ void embed_kernel(const int* __restrict__ tok,
                             const float* __restrict__ table,
                             const float* __restrict__ gemb,
                             float* __restrict__ x) {
    int row = blockIdx.x;                 // 0..BS_-1
    int b = row / S_, s = row % S_;
    const float* src = (s < SL_) ? (table + (size_t)tok[b * SL_ + s] * D_) : gemb;
    float* dst = x + (size_t)row * D_;
    for (int d = threadIdx.x; d < D_; d += blockDim.x) dst[d] = src[d];
}

// -------------------------------------------------------------------------
// Structured attention, long queries: each (b,h,seg) wave handles 16 queries
// x 17 keys (own segment + own global).  Lanes own dims {lane, lane+32}.
// -------------------------------------------------------------------------
__global__ void attn_long(const float* __restrict__ qkv,
                          const int* __restrict__ pad_long,
                          const int* __restrict__ pad_glob,
                          float* __restrict__ o) {
    const int id  = blockIdx.x;                    // b*H*SG + h*SG + seg
    const int seg = id % SG_;
    const int h   = (id / SG_) % H_;
    const int b   = id / (SG_ * H_);
    const int lane = threadIdx.x;
    const int d0 = lane, d1 = lane + 32;
    const float scale = 0.125f;                    // 1/sqrt(64)

    float kd0[17], kd1[17], vd0[17], vd1[17], keyb[17];
#pragma unroll
    for (int j = 0; j < 17; ++j) {
        int pos = (j < 16) ? (seg * RATIO_ + j) : (SL_ + seg);
        size_t base = ((size_t)(b * S_ + pos)) * (3 * D_) + h * HD_;
        kd0[j] = qkv[base + D_ + d0];
        kd1[j] = qkv[base + D_ + d1];
        vd0[j] = qkv[base + 2 * D_ + d0];
        vd1[j] = qkv[base + 2 * D_ + d1];
        int pad = (j < 16) ? pad_long[b * SL_ + seg * RATIO_ + j]
                           : pad_glob[b * SG_ + seg];
        keyb[j] = pad ? -1e9f : 0.0f;
    }

    for (int i = 0; i < 16; ++i) {
        int qpos = seg * RATIO_ + i;
        size_t qb = ((size_t)(b * S_ + qpos)) * (3 * D_) + h * HD_;
        float q0 = qkv[qb + d0], q1 = qkv[qb + d1];
        float sarr[17];
        float mx = -1e30f;
#pragma unroll
        for (int j = 0; j < 17; ++j) {
            float part = q0 * kd0[j] + q1 * kd1[j];
#pragma unroll
            for (int m = 16; m >= 1; m >>= 1) part += __shfl_xor(part, m, 32);
            float s = part * scale + keyb[j];
            sarr[j] = s;
            mx = fmaxf(mx, s);
        }
        float sum = 0.0f, o0 = 0.0f, o1 = 0.0f;
#pragma unroll
        for (int j = 0; j < 17; ++j) {
            float p = __expf(sarr[j] - mx);
            sum += p;
            o0 += p * vd0[j];
            o1 += p * vd1[j];
        }
        float inv = 1.0f / sum;
        size_t ob = ((size_t)(b * S_ + qpos)) * D_ + h * HD_;
        o[ob + d0] = o0 * inv;
        o[ob + d1] = o1 * inv;
    }
}

// Global queries: one wave per (b,h,g); 144 keys (16 own long + 128 globals).
// Two passes over keys (max, then exp/accumulate) to keep registers low.
__global__ void attn_glob(const float* __restrict__ qkv,
                          const int* __restrict__ pad_long,
                          const int* __restrict__ pad_glob,
                          float* __restrict__ o) {
    const int id  = blockIdx.x;                    // b*H*SG + h*SG + g
    const int g   = id % SG_;
    const int h   = (id / SG_) % H_;
    const int b   = id / (SG_ * H_);
    const int lane = threadIdx.x;
    const int d0 = lane, d1 = lane + 32;
    const float scale = 0.125f;

    const int qpos = SL_ + g;
    size_t qb = ((size_t)(b * S_ + qpos)) * (3 * D_) + h * HD_;
    const float q0 = qkv[qb + d0], q1 = qkv[qb + d1];

    float mx = -1e30f;
    for (int j = 0; j < 16 + SG_; ++j) {
        int pos;  float bv;
        if (j < 16) { pos = g * RATIO_ + j; bv = pad_long[b * SL_ + pos] ? -1e9f : 0.0f; }
        else        { int gg = j - 16; pos = SL_ + gg; bv = pad_glob[b * SG_ + gg] ? -1e9f : 0.0f; }
        size_t kb = ((size_t)(b * S_ + pos)) * (3 * D_) + h * HD_;
        float part = q0 * qkv[kb + D_ + d0] + q1 * qkv[kb + D_ + d1];
#pragma unroll
        for (int m = 16; m >= 1; m >>= 1) part += __shfl_xor(part, m, 32);
        mx = fmaxf(mx, part * scale + bv);
    }

    float sum = 0.0f, o0 = 0.0f, o1 = 0.0f;
    for (int j = 0; j < 16 + SG_; ++j) {
        int pos;  float bv;
        if (j < 16) { pos = g * RATIO_ + j; bv = pad_long[b * SL_ + pos] ? -1e9f : 0.0f; }
        else        { int gg = j - 16; pos = SL_ + gg; bv = pad_glob[b * SG_ + gg] ? -1e9f : 0.0f; }
        size_t kb = ((size_t)(b * S_ + pos)) * (3 * D_) + h * HD_;
        float part = q0 * qkv[kb + D_ + d0] + q1 * qkv[kb + D_ + d1];
#pragma unroll
        for (int m = 16; m >= 1; m >>= 1) part += __shfl_xor(part, m, 32);
        float p = __expf(part * scale + bv - mx);
        sum += p;
        o0 += p * qkv[kb + 2 * D_ + d0];
        o1 += p * qkv[kb + 2 * D_ + d1];
    }
    float inv = 1.0f / sum;
    size_t ob = ((size_t)(b * S_ + qpos)) * D_ + h * HD_;
    o[ob + d0] = o0 * inv;
    o[ob + d1] = o1 * inv;
}

// -------------------------------------------------------------------------
// x = LayerNorm(x + y) * gamma + beta   (one 128-thread block per row, D=512)
// -------------------------------------------------------------------------
__global__ void ln_residual(float* __restrict__ x,
                            const float* __restrict__ y,
                            const float* __restrict__ gamma,
                            const float* __restrict__ beta) {
    const int row = blockIdx.x;
    const int t = threadIdx.x;               // 128 threads
    __shared__ float sred[128];
    size_t base = (size_t)row * D_;

    float v[4];
#pragma unroll
    for (int j = 0; j < 4; ++j) {
        int d = t + j * 128;
        v[j] = x[base + d] + y[base + d];
    }
    float s = v[0] + v[1] + v[2] + v[3];
    sred[t] = s;
    __syncthreads();
    for (int off = 64; off > 0; off >>= 1) {
        if (t < off) sred[t] += sred[t + off];
        __syncthreads();
    }
    float mu = sred[0] / (float)D_;
    __syncthreads();

    float vs = 0.0f;
#pragma unroll
    for (int j = 0; j < 4; ++j) { float dlt = v[j] - mu; vs += dlt * dlt; }
    sred[t] = vs;
    __syncthreads();
    for (int off = 64; off > 0; off >>= 1) {
        if (t < off) sred[t] += sred[t + off];
        __syncthreads();
    }
    float rstd = rsqrtf(sred[0] / (float)D_ + 1e-5f);

#pragma unroll
    for (int j = 0; j < 4; ++j) {
        int d = t + j * 128;
        x[base + d] = (v[j] - mu) * rstd * gamma[d] + beta[d];
    }
}

// -------------------------------------------------------------------------
// Masked mean-pool over global tokens + 2-layer classifier.  Tiny: 1 block.
// -------------------------------------------------------------------------
__global__ void pool_cls(const float* __restrict__ x,
                         const int* __restrict__ pad_glob,
                         const float* __restrict__ w1,
                         const float* __restrict__ w2,
                         float* __restrict__ out) {
    __shared__ float z[B_][D_];
    __shared__ float h1[B_][D_ / 2];
    __shared__ float cnt[B_];
    const int t = threadIdx.x;               // 256 threads

    if (t < B_) {
        int c = 0;
        for (int g = 0; g < SG_; ++g) c += (pad_glob[t * SG_ + g] == 0);
        cnt[t] = (float)c;
    }
    __syncthreads();

    for (int idx = t; idx < B_ * D_; idx += blockDim.x) {
        int b = idx / D_, d = idx % D_;
        float s = 0.0f;
        for (int g = 0; g < SG_; ++g)
            if (!pad_glob[b * SG_ + g])
                s += x[((size_t)(b * S_ + SL_ + g)) * D_ + d];
        z[b][d] = s / cnt[b];
    }
    __syncthreads();

    for (int idx = t; idx < B_ * (D_ / 2); idx += blockDim.x) {
        int b = idx / (D_ / 2), j = idx % (D_ / 2);
        float s = 0.0f;
        for (int d = 0; d < D_; ++d) s += z[b][d] * w1[j * D_ + d];
        h1[b][j] = fmaxf(s, 0.0f);
    }
    __syncthreads();

    for (int idx = t; idx < B_ * NC_; idx += blockDim.x) {
        int b = idx / NC_, c = idx % NC_;
        float s = 0.0f;
        for (int j = 0; j < D_ / 2; ++j) s += h1[b][j] * w2[c * (D_ / 2) + j];
        out[idx] = s;
    }
}

// -------------------------------------------------------------------------
extern "C" void kernel_launch(void* const* d_in, const int* in_sizes, int n_in,
                              void* d_out, int out_size, void* d_ws, size_t ws_size,
                              hipStream_t stream) {
    const int*   token_ids   = (const int*)  d_in[0];
    const float* embed_table = (const float*)d_in[1];
    const float* global_emb  = (const float*)d_in[2];
    const float* in_proj_w   = (const float*)d_in[3];   // (L,3D,D)
    const float* in_proj_b   = (const float*)d_in[4];   // (L,3D)
    const float* out_w       = (const float*)d_in[5];   // (L,D,D)
    const float* out_b       = (const float*)d_in[6];   // (L,D)
    const float* ln1_g       = (const float*)d_in[7];
    const float* ln1_b       = (const float*)d_in[8];
    const float* ln2_g       = (const float*)d_in[9];
    const float* ln2_b       = (const float*)d_in[10];
    const float* ff1_w       = (const float*)d_in[11];  // (L,F,D)
    const float* ff1_b       = (const float*)d_in[12];  // (L,F)
    const float* ff2_w       = (const float*)d_in[13];  // (L,D,F)
    const float* ff2_b       = (const float*)d_in[14];  // (L,D)
    const float* cls_w1      = (const float*)d_in[15];  // (D/2,D)
    const float* cls_w2      = (const float*)d_in[16];  // (NC,D/2)

    // workspace carve-up (floats)
    float* ws   = (float*)d_ws;
    float* x    = ws;                                   // BS*D
    float* buf1 = x    + (size_t)BS_ * D_;              // BS*F (covers qkv & h)
    float* buf2 = buf1 + (size_t)BS_ * F_;              // BS*D
    int*   pad_long = (int*)(buf2 + (size_t)BS_ * D_);  // B*SL
    int*   pad_glob = pad_long + B_ * SL_;              // B*SG

    // masks + embedding
    mask_kernel<<<dim3((B_ * SG_ + 63) / 64), dim3(64), 0, stream>>>(
        token_ids, pad_long, pad_glob);
    embed_kernel<<<dim3(BS_), dim3(128), 0, stream>>>(
        token_ids, embed_table, global_emb, x);

    for (int i = 0; i < L_; ++i) {
        const float* wq  = in_proj_w + (size_t)i * 3 * D_ * D_;
        const float* bq  = in_proj_b + (size_t)i * 3 * D_;
        const float* wo  = out_w     + (size_t)i * D_ * D_;
        const float* bo  = out_b     + (size_t)i * D_;
        const float* w1  = ff1_w     + (size_t)i * F_ * D_;
        const float* b1  = ff1_b     + (size_t)i * F_;
        const float* w2  = ff2_w     + (size_t)i * D_ * F_;
        const float* b2  = ff2_b     + (size_t)i * D_;

        // qkv = x @ Wqkv^T + b      (BS x 3D)
        gemm_wmma<false><<<dim3(BS_ / 64, (3 * D_) / 64), dim3(128), 0, stream>>>(
            x, wq, bq, buf1, BS_, 3 * D_, D_);

        // structured attention -> buf2 (BS x D)
        attn_long<<<dim3(B_ * H_ * SG_), dim3(32), 0, stream>>>(
            buf1, pad_long, pad_glob, buf2);
        attn_glob<<<dim3(B_ * H_ * SG_), dim3(32), 0, stream>>>(
            buf1, pad_long, pad_glob, buf2);

        // out proj -> buf1 (qkv dead), then x = LN(x + buf1)
        gemm_wmma<false><<<dim3(BS_ / 64, D_ / 64), dim3(128), 0, stream>>>(
            buf2, wo, bo, buf1, BS_, D_, D_);
        ln_residual<<<dim3(BS_), dim3(128), 0, stream>>>(
            x, buf1, ln1_g + (size_t)i * D_, ln1_b + (size_t)i * D_);

        // FFN: h = relu(x @ W1^T + b1) -> buf1 (BS x F)
        gemm_wmma<true><<<dim3(BS_ / 64, F_ / 64), dim3(128), 0, stream>>>(
            x, w1, b1, buf1, BS_, F_, D_);
        // g = h @ W2^T + b2 -> buf2, then x = LN(x + buf2)
        gemm_wmma<false><<<dim3(BS_ / 64, D_ / 64), dim3(128), 0, stream>>>(
            buf1, w2, b2, buf2, BS_, D_, F_);
        ln_residual<<<dim3(BS_), dim3(128), 0, stream>>>(
            x, buf2, ln2_g + (size_t)i * D_, ln2_b + (size_t)i * D_);
    }

    // pool over globals + classifier
    pool_cls<<<dim3(1), dim3(256), 0, stream>>>(
        x, pad_glob, cls_w1, cls_w2, (float*)d_out);
}
